// TextCnn_1726576857496
// MI455X (gfx1250) — matrix operand order repaired and checked by reading, not compiled, not run on hardware
//
#include <hip/hip_runtime.h>
#include <hip/hip_bf16.h>
#include <stdint.h>

// ---------------------------------------------------------------------------
// TextCNN forward for MI455X (gfx1250, wave32, WMMA).
// Per masked-in token: causal convs (k=3,4,5) over previous <=5 embeddings
// -> relu -> 2304->768 linear -> LN -> +emb -> mask.
// Two bf16 WMMA GEMMs with 2-way M register blocking (1 b128 B-load per WMMA).
// ---------------------------------------------------------------------------

typedef __attribute__((ext_vector_type(16))) __bf16  v16bf;
typedef __attribute__((ext_vector_type(8)))  float   v8f;
typedef __attribute__((ext_vector_type(4)))  unsigned int u32x4;
typedef __attribute__((ext_vector_type(4)))  float   f32x4;

#define BSZ 64
#define TSZ 512
#define DSZ 768
#define C3  2304              // 3 * 768 concat channels
#define KW  3840              // 5 * 768 stacked-window K
#define NTOK (BSZ * TSZ)

#define ROWB 1552             // LDS row stride bytes: 768*2 payload + 16 pad
#define MT   128              // tokens per block in conv GEMM
#define WROWS (MT + 4)        // +4 history rows

// workspace layout (bytes)
#define WBIG_BYTES ((size_t)C3 * KW * 2)          // 17,694,720
#define WL_BYTES   ((size_t)DSZ * C3 * 2)         //  3,538,944
#define H_OFFSET   (WBIG_BYTES + WL_BYTES)

__device__ __forceinline__ unsigned short f2bf(float x) {
  unsigned int u = __float_as_uint(x);
  u += 0x7fffu + ((u >> 16) & 1u);          // round-to-nearest-even
  return (unsigned short)(u >> 16);
}
__device__ __forceinline__ unsigned int pack2(float a, float b) {
  return (unsigned int)f2bf(a) | ((unsigned int)f2bf(b) << 16);
}

union ABu { u32x4 u[2]; v16bf v; };

// ---------------------------------------------------------------------------
// Kernel 0: weight repack into bf16.
//   Wbig[c][p*768+d]: c<768 conv3 (blocks 1..3), c<1536 conv4 (blocks 0..3),
//   else conv5 (blocks 0..4); zero elsewhere. Row-major, K contiguous -> B frag.
//   WlB[d][c] = bf16(Wl) -- K(=c) contiguous per N(=d) row.
// ---------------------------------------------------------------------------
__global__ void k_prep(const float* __restrict__ W3, const float* __restrict__ W4,
                       const float* __restrict__ W5, const float* __restrict__ Wl,
                       unsigned short* __restrict__ wbig, unsigned short* __restrict__ wlb) {
  const size_t NW  = (size_t)C3 * KW;
  const size_t NL  = (size_t)DSZ * C3;
  const size_t tot = NW + NL;
  for (size_t i = (size_t)blockIdx.x * blockDim.x + threadIdx.x; i < tot;
       i += (size_t)gridDim.x * blockDim.x) {
    if (i < NW) {
      const int c = (int)(i / KW);
      const int k = (int)(i - (size_t)c * KW);
      const int p = k / DSZ;
      const int d = k - p * DSZ;
      float v = 0.f;
      if (c < 768) {
        if (p >= 1 && p <= 3) v = W3[((size_t)c * DSZ + d) * 3 + (p - 1)];
      } else if (c < 1536) {
        if (p < 4) v = W4[((size_t)(c - 768) * DSZ + d) * 4 + p];
      } else {
        v = W5[((size_t)(c - 1536) * DSZ + d) * 5 + p];
      }
      wbig[i] = f2bf(v);
    } else {
      const size_t q = i - NW;
      wlb[q] = f2bf(Wl[q]);
    }
  }
}

// ---------------------------------------------------------------------------
// Kernel 1: fused conv GEMM.  H[tok][c] = relu(Xwin . Wbig^T + bias), bf16 out.
// Block: 256 threads (8 waves), 128 tokens x full 2304 channels.
// LDS: 132 x 768 bf16 window (row stride 1552B), ~200 KB (<=320 KB/WGP).
// Wave w: M-pair wq=w&3 (two 16-token subtiles), N-half nh=w>>2 (72 N-tiles,
// 12 groups of 6). Each B fragment feeds 2 WMMAs (M blocking) and each A
// fragment feeds 6 -> 1 global b128 per WMMA.
// ---------------------------------------------------------------------------
__global__ __launch_bounds__(256) void k_conv_gemm(
    const float* __restrict__ emb, const unsigned short* __restrict__ wbig,
    const float* __restrict__ b3, const float* __restrict__ b4,
    const float* __restrict__ b5, unsigned short* __restrict__ H) {
  extern __shared__ char sm[];
  const int mb  = blockIdx.x;               // 256 token-blocks
  const int b   = mb >> 2;
  const int t0  = (mb & 3) << 7;
  const int tid = threadIdx.x;

  // Stage window tokens t0-4 .. t0+127 as bf16 (zeros for t<0).
  for (int i = tid; i < WROWS * 192; i += 256) {
    const int r = i / 192;
    const int q = i - r * 192;
    const int t = t0 - 4 + r;
    f32x4 v = {0.f, 0.f, 0.f, 0.f};
    if (t >= 0) v = *(const f32x4*)(emb + ((size_t)(b * TSZ + t)) * DSZ + q * 4);
    unsigned int* dst = (unsigned int*)(sm + (size_t)r * ROWB + q * 8);
    dst[0] = pack2(v.x, v.y);
    dst[1] = pack2(v.z, v.w);
  }
  __syncthreads();

  const int w    = tid >> 5;
  const int lane = tid & 31;
  const int lr   = lane & 15;
  const int h    = lane >> 4;
  const int wq   = w & 3;                   // which 32-token M-pair
  const int nh   = w >> 2;                  // which half of the 144 N-tiles
  // A fragment (16x32 bf16): lane row M = lr; b128 pair at K=8h / K=16+8h.
  const char* aBase0 = sm + (size_t)(32 * wq + lr) * ROWB + 16 * h;
  const char* aBase1 = aBase0 + (size_t)16 * ROWB;

  for (int g = 0; g < 12; ++g) {
    const int nt0 = nh * 72 + g * 6;        // first of 6 N-tiles this group
    const char* bp[6];
#pragma unroll
    for (int j = 0; j < 6; ++j)
      bp[j] = (const char*)wbig + ((size_t)((nt0 + j) * 16 + lr) * KW + 8 * h) * 2;

    v8f acc0[6] = {};
    v8f acc1[6] = {};
    for (int p = 0; p < 5; ++p) {           // window position (token t-4+p)
      const char* aRow0 = aBase0 + p * ROWB;
      const char* aRow1 = aBase1 + p * ROWB;
      for (int d0 = 0; d0 < DSZ; d0 += 32) {
        ABu a0, a1;
        a0.u[0] = *(const u32x4*)(aRow0 + d0 * 2);
        a0.u[1] = *(const u32x4*)(aRow0 + d0 * 2 + 32);
        a1.u[0] = *(const u32x4*)(aRow1 + d0 * 2);
        a1.u[1] = *(const u32x4*)(aRow1 + d0 * 2 + 32);
        const int kk2 = (p * DSZ + d0) * 2;
#pragma unroll
        for (int j = 0; j < 6; ++j) {
          ABu bb;
          const u32x4* q4 = (const u32x4*)(bp[j] + kk2);
          bb.u[0] = q4[0];
          bb.u[1] = q4[2];
          acc0[j] = __builtin_amdgcn_wmma_f32_16x16x32_bf16(
              false, a0.v, false, bb.v, (short)0, acc0[j], false, false);
          acc1[j] = __builtin_amdgcn_wmma_f32_16x16x32_bf16(
              false, a1.v, false, bb.v, (short)0, acc1[j], false, false);
        }
      }
    }
    // bias + relu + bf16 store (C/D layout: lane col N=lr, rows M=v+8h).
#pragma unroll
    for (int j = 0; j < 6; ++j) {
      const int c = (nt0 + j) * 16 + lr;
      const float bias = (c < 768) ? b3[c] : ((c < 1536) ? b4[c - 768] : b5[c - 1536]);
#pragma unroll
      for (int v = 0; v < 8; ++v) {
        float z0 = acc0[j][v] + bias;
        float z1 = acc1[j][v] + bias;
        z0 = z0 > 0.f ? z0 : 0.f;
        z1 = z1 > 0.f ? z1 : 0.f;
        const int tokA = t0 + 32 * wq + v + 8 * h;
        const int tokB = tokA + 16;
        H[(size_t)(b * TSZ + tokA) * C3 + c] = f2bf(z0);
        H[(size_t)(b * TSZ + tokB) * C3 + c] = f2bf(z1);
      }
    }
  }
}

// ---------------------------------------------------------------------------
// Kernel 2: tok = H . Wl^T + bl, LayerNorm over D, +emb, x mask.
// Block handles 32 tokens (two M-subtiles/wave; 512%32==0 so single batch).
// Wave w computes 6 of the 48 N-tiles for both M-subtiles (full K=2304);
// results land in a 32x768 LDS tile; 8 threads/token do the LN reduction.
// ---------------------------------------------------------------------------
__global__ __launch_bounds__(256) void k_lin_ln(
    const unsigned short* __restrict__ H, const unsigned short* __restrict__ wlb,
    const float* __restrict__ bl, const float* __restrict__ emb,
    const unsigned char* __restrict__ mask, float* __restrict__ out) {
  extern __shared__ float zs[];             // 32 x 772 floats
  const int tg0  = blockIdx.x * 32;
  const int tid  = threadIdx.x;
  const int w    = tid >> 5;
  const int lane = tid & 31;
  const int lr   = lane & 15;
  const int h    = lane >> 4;

  const char* aBase0 = (const char*)H + ((size_t)(tg0 + lr) * C3 + 8 * h) * 2;
  const char* aBase1 = aBase0 + (size_t)16 * C3 * 2;
  const char* bp[6];
#pragma unroll
  for (int i = 0; i < 6; ++i)
    bp[i] = (const char*)wlb + ((size_t)((w * 6 + i) * 16 + lr) * C3 + 8 * h) * 2;

  v8f acc0[6] = {};
  v8f acc1[6] = {};
  for (int kk = 0; kk < C3; kk += 32) {
    ABu a0, a1;
    a0.u[0] = *(const u32x4*)(aBase0 + kk * 2);
    a0.u[1] = *(const u32x4*)(aBase0 + kk * 2 + 32);
    a1.u[0] = *(const u32x4*)(aBase1 + kk * 2);
    a1.u[1] = *(const u32x4*)(aBase1 + kk * 2 + 32);
#pragma unroll
    for (int i = 0; i < 6; ++i) {
      ABu bb;
      const u32x4* q4 = (const u32x4*)(bp[i] + kk * 2);
      bb.u[0] = q4[0];
      bb.u[1] = q4[2];
      acc0[i] = __builtin_amdgcn_wmma_f32_16x16x32_bf16(
          false, a0.v, false, bb.v, (short)0, acc0[i], false, false);
      acc1[i] = __builtin_amdgcn_wmma_f32_16x16x32_bf16(
          false, a1.v, false, bb.v, (short)0, acc1[i], false, false);
    }
  }
#pragma unroll
  for (int i = 0; i < 6; ++i) {
    const int d = (w * 6 + i) * 16 + lr;
    const float bias = bl[d];
#pragma unroll
    for (int v = 0; v < 8; ++v) {
      zs[(v + 8 * h) * 772 + d]        = acc0[i][v] + bias;
      zs[(16 + v + 8 * h) * 772 + d]   = acc1[i][v] + bias;
    }
  }
  __syncthreads();

  const int m = tid >> 3;      // token row 0..31
  const int j = tid & 7;       // 8 threads cooperate per token
  float s = 0.f, ss = 0.f;
#pragma unroll
  for (int q = 0; q < 96; ++q) {
    const float z = zs[m * 772 + j + 8 * q];
    s += z;
    ss += z * z;
  }
#pragma unroll
  for (int o = 4; o >= 1; o >>= 1) {
    s  += __shfl_xor(s, o, 8);
    ss += __shfl_xor(ss, o, 8);
  }
  const float mean = s * (1.f / 768.f);
  const float var  = ss * (1.f / 768.f) - mean * mean;
  const float rs   = rsqrtf(var + 1e-5f);

  const int n  = tg0 + m;
  const int bb = n >> 9;
  const int t  = n & 511;
  const float mk = mask[bb * TSZ + t] ? 1.f : 0.f;
  const float* ep = emb + (size_t)n * DSZ;
  float* op = out + (size_t)n * DSZ;
  for (int q = 0; q < 96; ++q) {
    const int d = j + 8 * q;
    const float z = zs[m * 772 + d];
    op[d] = mk * (ep[d] + (z - mean) * rs);
  }
}

// ---------------------------------------------------------------------------
extern "C" void kernel_launch(void* const* d_in, const int* in_sizes, int n_in,
                              void* d_out, int out_size, void* d_ws, size_t ws_size,
                              hipStream_t stream) {
  (void)in_sizes; (void)n_in; (void)out_size; (void)ws_size;
  const float*         emb  = (const float*)d_in[0];
  const unsigned char* mask = (const unsigned char*)d_in[1];
  const float* W3 = (const float*)d_in[2];
  const float* b3 = (const float*)d_in[3];
  const float* W4 = (const float*)d_in[4];
  const float* b4 = (const float*)d_in[5];
  const float* W5 = (const float*)d_in[6];
  const float* b5 = (const float*)d_in[7];
  const float* Wl = (const float*)d_in[8];
  const float* bl = (const float*)d_in[9];

  unsigned short* wbig = (unsigned short*)d_ws;
  unsigned short* wlb  = (unsigned short*)((char*)d_ws + WBIG_BYTES);
  unsigned short* H    = (unsigned short*)((char*)d_ws + H_OFFSET);
  float* out = (float*)d_out;

  k_prep<<<2048, 256, 0, stream>>>(W3, W4, W5, Wl, wbig, wlb);

  const int smem1 = WROWS * ROWB;  // 204,864 B dynamic LDS (<=320KB/WGP on CDNA5)
  hipFuncSetAttribute(reinterpret_cast<const void*>(k_conv_gemm),
                      hipFuncAttributeMaxDynamicSharedMemorySize, smem1);
  k_conv_gemm<<<256, 256, smem1, stream>>>(emb, wbig, b3, b4, b5, H);

  const int smem2 = 32 * 772 * 4;  // 98,816 B dynamic LDS
  hipFuncSetAttribute(reinterpret_cast<const void*>(k_lin_ln),
                      hipFuncAttributeMaxDynamicSharedMemorySize, smem2);
  k_lin_ln<<<NTOK / 32, 256, smem2, stream>>>(H, wlb, bl, emb, mask, out);
}
